// Seq2Seq_37898791420607
// MI455X (gfx1250) — matrix operand order repaired and checked by reading, not compiled, not run on hardware
//
#include <hip/hip_runtime.h>
#include <hip/hip_bf16.h>
#include <math.h>

typedef __bf16 bhalf;
typedef __attribute__((ext_vector_type(16))) __bf16 v16bf;
typedef __attribute__((ext_vector_type(8)))  float  v8f;
typedef __attribute__((ext_vector_type(4)))  unsigned int v4u;
typedef __attribute__((ext_vector_type(8)))  int v8i;
typedef __attribute__((ext_vector_type(4)))  int v4i;

#if defined(__AMDGCN__) && defined(__has_builtin)
#if __has_builtin(__builtin_amdgcn_tensor_load_to_lds) && __has_builtin(__builtin_amdgcn_s_wait_tensorcnt)
#define GEMM_TDM 1
#endif
#endif

__device__ __forceinline__ float sigm_(float x) { return 1.0f / (1.0f + expf(-x)); }
__device__ __forceinline__ float gelu_(float x) { return 0.5f * x * (1.0f + erff(x * 0.70710678118654752f)); }

// Pack TDM descriptor group0: count=1, lds_addr, 57-bit global addr, type=2.
__device__ __forceinline__ v4u tdm_g0(unsigned lds, const void* gaddr) {
  unsigned long long ga = (unsigned long long)gaddr;
  v4u g;
  g.x = 1u;                                   // count=1 (valid descriptor)
  g.y = lds;                                  // LDS byte address
  g.z = (unsigned)ga;                         // global_addr[31:0]
  g.w = (unsigned)((ga >> 32) & 0x1FFFFFFu) | (2u << 30);  // addr[56:32] | type=2
  return g;
}

// ---------------------------------------------------------------------------
// bf16 WMMA GEMM:  C[M,N] = act(A[M,K] @ B[K,N] + bias[N])
// Workgroup = WM x WN waves; each wave computes one 16x64 tile (4 WMMA frags).
// Requires: M % (16*WM) == 0, N % (64*WN) == 0, K % 32 == 0.
// A/B tiles staged to LDS by the Tensor Data Mover with LDS row padding
// (+8 bf16/row) applied by the TDM pad engine; double-buffered so the DMA of
// tile k+1 overlaps WMMA on tile k.  OUTMODE: 0=f32, 1=bf16, 2=both.
// SCATTER=1: row r -> out row (b*64 + t + 1), b=r&31, t=r>>5.
// ---------------------------------------------------------------------------
template <int WM, int WN, int ACT, int SCATTER, int OUTMODE>
__global__ __launch_bounds__(WM * WN * 32)
void gemm_bf16_wmma(const bhalf* __restrict__ A, const bhalf* __restrict__ B,
                    const float* __restrict__ bias,
                    float* __restrict__ Cf, bhalf* __restrict__ Cb,
                    int M, int N, int K) {
  constexpr int TM = WM * 16;
  constexpr int TN = WN * 64;
  __shared__ __align__(16) bhalf As[2][TM][32 + 8];
  __shared__ __align__(16) bhalf Bs[2][32][TN + 8];

  const int tid  = threadIdx.x;
  const int lane = tid & 31;
  const int w    = tid >> 5;
  const int wm   = w / WN;
  const int wn   = w % WN;
  const int half = lane >> 4;
  const int ml   = lane & 15;
  const int nthr = WM * WN * 32;
  const int rowBase = blockIdx.y * TM;
  const int colBase = blockIdx.x * TN;
  const int nkt = K >> 5;

  v8f acc[4];
#pragma unroll
  for (int f = 0; f < 4; ++f)
#pragma unroll
    for (int e = 0; e < 8; ++e) acc[f][e] = 0.0f;

#ifdef GEMM_TDM
  // pad_interval code: pad after 2^(c+1) dwords == one tile row; amount code 3 = 4 dwords (8 bf16)
  constexpr int AINT = 3;                       // A row: 32 bf16 = 16 dwords
  constexpr int BINT = (TN == 256) ? 6 : 4;     // B row: TN bf16 = TN/2 dwords
  constexpr unsigned PADW = (1u << 20) | (3u << 25);
  const v8i g1a = { (int)((1u << 16) | PADW | (AINT << 22)),
                    0, 16, 16 | (32 << 16), TM, K, 0, 0 };
  const v8i g1b = { (int)((1u << 16) | PADW | (BINT << 22)),
                    0, 16, 16 | (TN << 16), 32, N, 0, 0 };
  const v4i z4 = { 0, 0, 0, 0 };
  const v8i z8 = { 0, 0, 0, 0, 0, 0, 0, 0 };
  auto issue_tile = [&](int kt, int buf) {
    __builtin_amdgcn_tensor_load_to_lds(
        tdm_g0((unsigned)(size_t)&As[buf][0][0], &A[(size_t)rowBase * K + kt * 32]),
        g1a, z4, z4, z8, 0);
    __builtin_amdgcn_tensor_load_to_lds(
        tdm_g0((unsigned)(size_t)&Bs[buf][0][0], &B[(size_t)(kt * 32) * N + colBase]),
        g1b, z4, z4, z8, 0);
  };
  if (w == 0) issue_tile(0, 0);
#endif

  for (int kt = 0; kt < nkt; ++kt) {
    int cur;
#ifdef GEMM_TDM
    cur = kt & 1;
    if (w == 0) {
      if (kt + 1 < nkt) {
        issue_tile(kt + 1, cur ^ 1);                  // prefetch next tile
        __builtin_amdgcn_s_wait_tensorcnt(2);         // current tile's 2 DMAs done
      } else {
        __builtin_amdgcn_s_wait_tensorcnt(0);
      }
    }
    __syncthreads();
#else
    cur = 0;
    for (int i = tid; i < TM * 4; i += nthr) {
      int r = i >> 2, q = i & 3;
      *(uint4*)&As[0][r][q * 8] =
          *(const uint4*)&A[(size_t)(rowBase + r) * K + kt * 32 + q * 8];
    }
    for (int i = tid; i < 32 * (TN / 8); i += nthr) {
      int r = i / (TN / 8), q = i % (TN / 8);
      *(uint4*)&Bs[0][r][q * 8] =
          *(const uint4*)&B[(size_t)(kt * 32 + r) * N + colBase + q * 8];
    }
    __syncthreads();
#endif

    // A fragment: lane (0-15|16-31) holds row m; VGPR r holds K pair
    //   K = half*8 + (r&4 ? 16 : 0) + 2*(r&3)
    union { v16bf v; unsigned u[8]; } af;
    const int mrow = wm * 16 + ml;
#pragma unroll
    for (int r = 0; r < 8; ++r) {
      int k2 = half * 8 + ((r & 4) ? 16 : 0) + 2 * (r & 3);
      af.u[r] = *(const unsigned*)&As[cur][mrow][k2];
    }
    // B fragments: lane holds column n = lane&15; element s holds K = half*8 + (s&7) + (s&8 ? 16 : 0)
#pragma unroll
    for (int f = 0; f < 4; ++f) {
      union { v16bf v; bhalf e[16]; } bfr;
      const int bcol = wn * 64 + f * 16 + ml;
#pragma unroll
      for (int s = 0; s < 16; ++s)
        bfr.e[s] = Bs[cur][half * 8 + (s & 7) + ((s & 8) ? 16 : 0)][bcol];
      acc[f] = __builtin_amdgcn_wmma_f32_16x16x32_bf16(
          false, af.v, false, bfr.v, (short)0, acc[f], false, false);
    }
    __syncthreads();
  }

  // C layout: lane (0-15|16-31), VGPR v -> M = half*8 + v, N = lane&15
#pragma unroll
  for (int f = 0; f < 4; ++f) {
    int n = colBase + wn * 64 + f * 16 + ml;
    float bv = bias[n];
#pragma unroll
    for (int v = 0; v < 8; ++v) {
      int gr = rowBase + wm * 16 + half * 8 + v;
      float x = acc[f][v] + bv;
      if (ACT == 1) x = gelu_(x);
      size_t offo;
      if (SCATTER) {
        int b = gr & 31, t = gr >> 5;
        offo = (size_t)(b * 64 + t + 1) * (size_t)N + n;
      } else {
        offo = (size_t)gr * (size_t)N + n;
      }
      if (OUTMODE == 0 || OUTMODE == 2) Cf[offo] = x;
      if (OUTMODE == 1 || OUTMODE == 2) Cb[offo] = (bhalf)x;
    }
  }
}

// ---------------------------------------------------------------------------
// Small helper kernels
// ---------------------------------------------------------------------------
__global__ void k_f32_to_bf16(const float* __restrict__ in, bhalf* __restrict__ out, int n) {
  int i = blockIdx.x * blockDim.x + threadIdx.x;
  if (i < n) out[i] = (bhalf)in[i];
}
__global__ void k_zero_f32(float* p, int n) {
  int i = blockIdx.x * blockDim.x + threadIdx.x;
  if (i < n) p[i] = 0.0f;
}
__global__ void k_zero_bf16(bhalf* p, int n) {
  int i = blockIdx.x * blockDim.x + threadIdx.x;
  if (i < n) p[i] = (bhalf)0.0f;
}
__global__ void k_zero_out_row0(float* out) {  // d_out[:,0,:] = 0
  int i = blockIdx.x * blockDim.x + threadIdx.x;
  if (i < 32 * 32000) {
    int b = i / 32000, v = i % 32000;
    out[(size_t)b * 64 * 32000 + v] = 0.0f;
  }
}
__global__ void k_gather_embed(const float* __restrict__ tab, const int* __restrict__ idx,
                               bhalf* __restrict__ out, int rows, int E) {
  int i = blockIdx.x * blockDim.x + threadIdx.x;
  if (i >= rows * E) return;
  int r = i / E, e = i - r * E;
  out[i] = (bhalf)tab[(size_t)idx[r] * E + e];
}
__global__ void k_gather_embed_dec(const float* __restrict__ tab, const int* __restrict__ trg,
                                   bhalf* __restrict__ out, int E) {
  int i = blockIdx.x * blockDim.x + threadIdx.x;
  if (i >= 63 * 32 * E) return;
  int r = i / E, e = i - r * E;
  int t = r >> 5, b = r & 31;
  out[i] = (bhalf)tab[(size_t)trg[b * 64 + t] * E + e];
}
__global__ void k_gru_gate_enc(const float* __restrict__ gi, const float* __restrict__ gh,
                               float* __restrict__ h, bhalf* __restrict__ hbf,
                               bhalf* __restrict__ wordreps,
                               const int* __restrict__ lens, int t, int col_base) {
  int i = blockIdx.x * blockDim.x + threadIdx.x;
  if (i >= 32 * 512) return;
  int b = i >> 9, j = i & 511;
  const float* gir = gi + (size_t)(b * 64 + t) * 1536;
  const float* ghr = gh + (size_t)b * 1536;
  float r = sigm_(gir[j] + ghr[j]);
  float z = sigm_(gir[512 + j] + ghr[512 + j]);
  float n = tanhf(gir[1024 + j] + r * ghr[1024 + j]);
  float hp = h[i];
  float hn = (1.0f - z) * n + z * hp;
  int L = lens[b]; if (L < 1) L = 1;
  bool valid = t < L;
  float ho = valid ? hn : hp;
  h[i] = ho;
  hbf[i] = (bhalf)ho;
  wordreps[(size_t)(b * 64 + t) * 1024 + col_base + j] = valid ? (bhalf)hn : (bhalf)0.0f;
}
__global__ void k_gru_gate_dec(const float* __restrict__ ge, const float* __restrict__ gh,
                               const float* __restrict__ hprev,
                               float* __restrict__ hnew, bhalf* __restrict__ hnew_bf, int t) {
  int i = blockIdx.x * blockDim.x + threadIdx.x;
  if (i >= 32 * 512) return;
  int b = i >> 9, j = i & 511;
  const float* gir = ge + (size_t)(t * 32 + b) * 1536;
  const float* ghr = gh + (size_t)b * 1536;
  float r = sigm_(gir[j] + ghr[j]);
  float z = sigm_(gir[512 + j] + ghr[512 + j]);
  float n = tanhf(gir[1024 + j] + r * ghr[1024 + j]);
  float hp = hprev[i];
  float hn = (1.0f - z) * n + z * hp;
  hnew[i] = hn;
  hnew_bf[i] = (bhalf)hn;
}
__global__ void k_pack_sentrep(const bhalf* __restrict__ hf, const bhalf* __restrict__ hb,
                               bhalf* __restrict__ out) {
  int i = blockIdx.x * blockDim.x + threadIdx.x;
  if (i >= 32 * 1024) return;
  int b = i >> 10, j = i & 1023;
  out[i] = (j < 512) ? hf[b * 512 + j] : hb[b * 512 + j - 512];
}
__global__ void k_attention_step(const float* __restrict__ q, const float* __restrict__ Kb,
                                 const float* __restrict__ Vb, const float* __restrict__ hnew,
                                 float* __restrict__ hdec_f, bhalf* __restrict__ hdec_bf,
                                 bhalf* __restrict__ hid_all, int t) {
  __shared__ float sc[64];
  __shared__ float al[64];
  int b = blockIdx.x;
  int tid = threadIdx.x;
  if (tid < 64) {
    float s = 0.0f;
    const float* qp = &q[b * 64];
    const float* kp = &Kb[((size_t)b * 64 + tid) * 64];
    for (int k = 0; k < 64; ++k) s += qp[k] * kp[k];
    sc[tid] = s * 0.125f;  // 1/sqrt(64)
  }
  __syncthreads();
  float m = -1e30f;
  for (int s = 0; s < 64; ++s) m = fmaxf(m, sc[s]);
  float den = 0.0f;
  for (int s = 0; s < 64; ++s) den += expf(sc[s] - m);
  if (tid < 64) al[tid] = expf(sc[tid] - m) / den;
  __syncthreads();
  for (int d = tid; d < 512; d += blockDim.x) {
    float a = 0.0f;
    for (int s = 0; s < 64; ++s) a += al[s] * Vb[((size_t)b * 64 + s) * 512 + d];
    float hid = hnew[b * 512 + d] + a;
    hdec_f[b * 512 + d] = hid;
    hdec_bf[b * 512 + d] = (bhalf)hid;
    hid_all[((size_t)t * 32 + b) * 512 + d] = (bhalf)hid;
  }
}

// ---------------------------------------------------------------------------
extern "C" void kernel_launch(void* const* d_in, const int* in_sizes, int n_in,
                              void* d_out, int out_size, void* d_ws, size_t ws_size,
                              hipStream_t stream) {
  const int*   src       = (const int*)d_in[0];
  const int*   trg       = (const int*)d_in[1];
  const int*   src_lens  = (const int*)d_in[2];
  const float* enc_embed = (const float*)d_in[3];
  const float* Wi_f = (const float*)d_in[4];
  const float* Wh_f = (const float*)d_in[5];
  const float* bi_f = (const float*)d_in[6];
  const float* bh_f = (const float*)d_in[7];
  const float* Wi_b = (const float*)d_in[8];
  const float* Wh_b = (const float*)d_in[9];
  const float* bi_b = (const float*)d_in[10];
  const float* bh_b = (const float*)d_in[11];
  const float* W_e2d = (const float*)d_in[12];
  const float* b_e2d = (const float*)d_in[13];
  const float* dec_embed = (const float*)d_in[14];
  const float* Wi_d = (const float*)d_in[15];
  const float* Wh_d = (const float*)d_in[16];
  const float* bi_d = (const float*)d_in[17];
  const float* bh_d = (const float*)d_in[18];
  const float* Wq = (const float*)d_in[19];
  const float* bq = (const float*)d_in[20];
  const float* Wk = (const float*)d_in[21];
  const float* bk = (const float*)d_in[22];
  const float* Wv = (const float*)d_in[23];
  const float* bv = (const float*)d_in[24];
  const float* W1 = (const float*)d_in[25];
  const float* b1 = (const float*)d_in[26];
  const float* W2 = (const float*)d_in[27];
  const float* b2 = (const float*)d_in[28];
  float* out = (float*)d_out;

  size_t off = 0;
  auto alloc = [&](size_t elems, size_t esz) -> void* {
    void* p = (char*)d_ws + off;
    off = (off + elems * esz + 255) & ~(size_t)255;
    return p;
  };
  bhalf* Wi_f_bf  = (bhalf*)alloc(256 * 1536, 2);
  bhalf* Wh_f_bf  = (bhalf*)alloc(512 * 1536, 2);
  bhalf* Wi_b_bf  = (bhalf*)alloc(256 * 1536, 2);
  bhalf* Wh_b_bf  = (bhalf*)alloc(512 * 1536, 2);
  bhalf* We2d_bf  = (bhalf*)alloc(1024 * 512, 2);
  bhalf* Wi_d_bf  = (bhalf*)alloc(256 * 1536, 2);
  bhalf* Wh_d_bf  = (bhalf*)alloc(512 * 1536, 2);
  bhalf* Wq_bf    = (bhalf*)alloc(512 * 64, 2);
  bhalf* Wk_bf    = (bhalf*)alloc(1024 * 64, 2);
  bhalf* Wv_bf    = (bhalf*)alloc(1024 * 512, 2);
  bhalf* W1_bf    = (bhalf*)alloc(512 * 512, 2);
  bhalf* W2_bf    = (bhalf*)alloc((size_t)512 * 32000, 2);
  bhalf* emb_bf   = (bhalf*)alloc(2048 * 256, 2);
  float* gi_f     = (float*)alloc((size_t)2048 * 1536, 4);
  float* gi_b     = (float*)alloc((size_t)2048 * 1536, 4);
  float* ge_d     = (float*)alloc((size_t)2016 * 1536, 4);
  bhalf* embd_bf  = (bhalf*)alloc(2016 * 256, 2);
  bhalf* wordreps = (bhalf*)alloc((size_t)2048 * 1024, 2);
  float* h_f      = (float*)alloc(32 * 512, 4);
  bhalf* h_f_bf   = (bhalf*)alloc(32 * 512, 2);
  float* h_b      = (float*)alloc(32 * 512, 4);
  bhalf* h_b_bf   = (bhalf*)alloc(32 * 512, 2);
  float* gh_buf   = (float*)alloc(32 * 1536, 4);
  bhalf* sent_bf  = (bhalf*)alloc(32 * 1024, 2);
  float* hdec_f   = (float*)alloc(32 * 512, 4);
  bhalf* hdec_bf  = (bhalf*)alloc(32 * 512, 2);
  float* hnew_f   = (float*)alloc(32 * 512, 4);
  bhalf* hnew_bf  = (bhalf*)alloc(32 * 512, 2);
  float* Kbuf     = (float*)alloc((size_t)2048 * 64, 4);
  float* Vbuf     = (float*)alloc((size_t)2048 * 512, 4);
  float* qbuf     = (float*)alloc(32 * 64, 4);
  bhalf* hid_all  = (bhalf*)alloc((size_t)2016 * 512, 2);
  bhalf* t1_bf    = (bhalf*)alloc((size_t)2016 * 512, 2);

  auto conv = [&](const float* s, bhalf* d, int n) {
    k_f32_to_bf16<<<(n + 255) / 256, 256, 0, stream>>>(s, d, n);
  };
  conv(Wi_f, Wi_f_bf, 256 * 1536);  conv(Wh_f, Wh_f_bf, 512 * 1536);
  conv(Wi_b, Wi_b_bf, 256 * 1536);  conv(Wh_b, Wh_b_bf, 512 * 1536);
  conv(W_e2d, We2d_bf, 1024 * 512); conv(Wi_d, Wi_d_bf, 256 * 1536);
  conv(Wh_d, Wh_d_bf, 512 * 1536);  conv(Wq, Wq_bf, 512 * 64);
  conv(Wk, Wk_bf, 1024 * 64);       conv(Wv, Wv_bf, 1024 * 512);
  conv(W1, W1_bf, 512 * 512);       conv(W2, W2_bf, 512 * 32000);

  auto G24 = [&](const bhalf* A, const bhalf* B, const float* bias,
                 float* Cf, int M, int N, int K) {
    dim3 g(N / 256, M / 32);
    gemm_bf16_wmma<2, 4, 0, 0, 0><<<g, 256, 0, stream>>>(A, B, bias, Cf, nullptr, M, N, K);
  };
  auto G21 = [&](const bhalf* A, const bhalf* B, const float* bias,
                 float* Cf, int M, int N, int K) {
    dim3 g(N / 64, M / 32);
    gemm_bf16_wmma<2, 1, 0, 0, 0><<<g, 64, 0, stream>>>(A, B, bias, Cf, nullptr, M, N, K);
  };

  // ---- encoder ----
  k_gather_embed<<<(2048 * 256 + 255) / 256, 256, 0, stream>>>(enc_embed, src, emb_bf, 2048, 256);
  G24(emb_bf, Wi_f_bf, bi_f, gi_f, 2048, 1536, 256);
  G24(emb_bf, Wi_b_bf, bi_b, gi_b, 2048, 1536, 256);
  k_zero_f32<<<64, 256, 0, stream>>>(h_f, 32 * 512);
  k_zero_bf16<<<64, 256, 0, stream>>>(h_f_bf, 32 * 512);
  k_zero_f32<<<64, 256, 0, stream>>>(h_b, 32 * 512);
  k_zero_bf16<<<64, 256, 0, stream>>>(h_b_bf, 32 * 512);
  for (int t = 0; t < 64; ++t) {
    G21(h_f_bf, Wh_f_bf, bh_f, gh_buf, 32, 1536, 512);
    k_gru_gate_enc<<<64, 256, 0, stream>>>(gi_f, gh_buf, h_f, h_f_bf, wordreps, src_lens, t, 0);
  }
  for (int t = 63; t >= 0; --t) {
    G21(h_b_bf, Wh_b_bf, bh_b, gh_buf, 32, 1536, 512);
    k_gru_gate_enc<<<64, 256, 0, stream>>>(gi_b, gh_buf, h_b, h_b_bf, wordreps, src_lens, t, 512);
  }

  // ---- bridge: hidden0 (gelu, f32+bf16 out), K, V ----
  k_pack_sentrep<<<(32 * 1024 + 255) / 256, 256, 0, stream>>>(h_f_bf, h_b_bf, sent_bf);
  {
    dim3 g(512 / 256, 1);
    gemm_bf16_wmma<2, 4, 1, 0, 2><<<g, 256, 0, stream>>>(sent_bf, We2d_bf, b_e2d,
                                                         hdec_f, hdec_bf, 32, 512, 1024);
  }
  G21(wordreps, Wk_bf, bk, Kbuf, 2048, 64, 1024);
  G24(wordreps, Wv_bf, bv, Vbuf, 2048, 512, 1024);

  // ---- decoder (sequential part: GRU + attention only) ----
  k_gather_embed_dec<<<(2016 * 256 + 255) / 256, 256, 0, stream>>>(dec_embed, trg, embd_bf, 256);
  G24(embd_bf, Wi_d_bf, bi_d, ge_d, 2016, 1536, 256);
  for (int t = 0; t < 63; ++t) {
    G21(hdec_bf, Wh_d_bf, bh_d, gh_buf, 32, 1536, 512);
    k_gru_gate_dec<<<64, 256, 0, stream>>>(ge_d, gh_buf, hdec_f, hnew_f, hnew_bf, t);
    G21(hnew_bf, Wq_bf, bq, qbuf, 32, 64, 512);
    k_attention_step<<<32, 256, 0, stream>>>(qbuf, Kbuf, Vbuf, hnew_f, hdec_f, hdec_bf, hid_all, t);
  }

  // ---- batched output projection: gelu(hid@W1+b1) (bf16) then @W2+b2 -> d_out rows 1..63 ----
  {
    dim3 g(512 / 256, 2016 / 32);
    gemm_bf16_wmma<2, 4, 1, 0, 1><<<g, 256, 0, stream>>>(hid_all, W1_bf, b1,
                                                         nullptr, t1_bf, 2016, 512, 512);
  }
  {
    dim3 g(32000 / 256, 2016 / 32);
    gemm_bf16_wmma<2, 4, 0, 1, 0><<<g, 256, 0, stream>>>(t1_bf, W2_bf, b2,
                                                         out, nullptr, 2016, 32000, 512);
  }
  k_zero_out_row0<<<(32 * 32000 + 255) / 256, 256, 0, stream>>>(out);
}
